// Model_33715493273856
// MI455X (gfx1250) — compile-verified, matrix operand order checked by reading, not run on hardware
//
#include <hip/hip_runtime.h>
#include <hip/hip_bf16.h>

// ---------------------------------------------------------------------------
// GraphSAGE 3-layer + global mean pool + FC for MI455X (gfx1250, wave32).
// GEMMs: LDS-staged bf16 A-panel (converted+swizzled once per block), all 8
// waves consume it via ds_load; v_wmma_f32_16x16x32_bf16 with f32 accumulate.
// Edge aggregation: float4 gather + f32 global atomics (agg fits in 192MB L2).
// ---------------------------------------------------------------------------

typedef __attribute__((ext_vector_type(16))) __bf16 v16bf;
typedef __attribute__((ext_vector_type(8)))  float  v8f;

#define NNODES 100000
#define NEDGES 1600000
#define HDIM   128
#define GRAPHS 64
#define NCLS   10

// ---------------- degree + scalar layer-1 aggregation ----------------------
__global__ void deg_agg1_kernel(const int* __restrict__ src, const int* __restrict__ dst,
                                const float* __restrict__ x,
                                float* __restrict__ cnt, float* __restrict__ agg1) {
    int e = blockIdx.x * blockDim.x + threadIdx.x;
    if (e >= NEDGES) return;
    int d = dst[e];
    atomicAdd(&cnt[d], 1.0f);
    atomicAdd(&agg1[d], x[src[e]]);
}

// ---------------- layer 1: rank-1 expansion [N,1] -> [N,128], ReLU ---------
__global__ void layer1_kernel(const float* __restrict__ x, const float* __restrict__ agg1,
                              const float* __restrict__ cnt,
                              const float* __restrict__ W1l, const float* __restrict__ b1,
                              const float* __restrict__ W1r,
                              float* __restrict__ h1) {
    unsigned t = blockIdx.x * blockDim.x + threadIdx.x;   // N*128 threads
    unsigned i = t >> 7;
    unsigned c = t & 127u;
    if (i >= NNODES) return;
    float m = agg1[i] / fmaxf(cnt[i], 1.0f);
    float v = m * W1l[c] + b1[c] + x[i] * W1r[c];
    h1[t] = v > 0.0f ? v : 0.0f;
}

// ---------------- edge aggregation: agg[dst,:] += h[src,:] (float4) --------
template <int LOGC>  // C channels, LOGC-2 float4 groups per edge
__global__ void edge_agg_kernel(const int* __restrict__ src, const int* __restrict__ dst,
                                const float* __restrict__ h, float* __restrict__ agg) {
    unsigned t = blockIdx.x * blockDim.x + threadIdx.x;   // E * C/4 threads
    unsigned e = t >> (LOGC - 2);
    unsigned c4 = (t & ((1u << (LOGC - 2)) - 1u)) << 2;   // channel base (x4)
    if (e >= NEDGES) return;
    unsigned s = (unsigned)src[e];
    unsigned d = (unsigned)dst[e];
    const float4 v = *(const float4*)(h + ((size_t)s << LOGC) + c4);
    float* a = agg + ((size_t)d << LOGC) + c4;
    atomicAdd(a + 0, v.x);
    atomicAdd(a + 1, v.y);
    atomicAdd(a + 2, v.z);
    atomicAdd(a + 3, v.w);
}

// ---------------- in-place mean: agg /= max(cnt,1) -------------------------
template <int LOGC>
__global__ void mean_scale_kernel(float* __restrict__ agg, const float* __restrict__ cnt) {
    unsigned t = blockIdx.x * blockDim.x + threadIdx.x;   // N*C threads
    unsigned i = t >> LOGC;
    if (i >= NNODES) return;
    agg[t] = agg[t] / fmaxf(cnt[i], 1.0f);
}

// ---------------- pack [Wl;Wr] (f32) into per-lane bf16 WMMA-B layout ------
// Wpack[((kt*NT + nt)*32 + lane)*16 + j], klocal = (j<8 ? j : j+8) + (lane>=16 ? 8:0)
template <int C, int COUT>
__global__ void pack_w_kernel(const float* __restrict__ Wl, const float* __restrict__ Wr,
                              __bf16* __restrict__ Wp) {
    unsigned idx = blockIdx.x * blockDim.x + threadIdx.x; // 2*C*COUT threads
    if (idx >= (unsigned)(2 * C * COUT)) return;
    unsigned j    = idx & 15u;
    unsigned lane = (idx >> 4) & 31u;
    unsigned tile = idx >> 9;
    const unsigned NT = COUT / 16;
    unsigned kt  = tile / NT;
    unsigned nt  = tile % NT;
    unsigned col = nt * 16 + (lane & 15u);
    unsigned k   = kt * 32 + ((lane >> 4) ? 8u : 0u) + (j < 8 ? j : j + 8);
    float v = (k < (unsigned)C) ? Wl[k * COUT + col] : Wr[(k - C) * COUT + col];
    Wp[idx] = (__bf16)v;
}

// ---------------- fused concat GEMM: out = relu([Amean|Aself] @ Wp + b) ----
// K = 2*C. Per block: the 16-row x K bf16 A-panel is staged ONCE into LDS in
// WMMA-A per-lane order; then 8 waves (one 16x16 col tile each) loop K doing
// ds_load(A) + global_load(B) + v_wmma_f32_16x16x32_bf16.
template <int C, int COUT>
__global__ void sage_gemm_kernel(const float* __restrict__ Amean,
                                 const float* __restrict__ Aself,
                                 const __bf16* __restrict__ Wp,
                                 const float* __restrict__ bias,
                                 float* __restrict__ out) {
    const int K  = 2 * C;
    const int KT = K / 32;            // K tiles
    const int NT = COUT / 16;         // column tiles total
    __shared__ alignas(32) __bf16 sA[KT * 32 * 16];   // [kt][lane][16] bf16

    const int tid  = threadIdx.x;
    const int lane = tid & 31;
    const int wave = tid >> 5;
    const int mTile = blockIdx.x * 16;

    // ---- phase 1: cooperative stage of A panel (f32 -> bf16, swizzled) ----
    // pair p = (kt, lane'); each thread builds one 32B v16bf entry.
#pragma unroll
    for (int p = tid; p < KT * 32; p += 256) {
        int kt  = p >> 5;
        int ln  = p & 31;
        int r   = ln & 15;
        int kb  = kt * 32 + ((ln >> 4) ? 8 : 0);
        const float* srcA;
        int kcol;
        if (kb < C) { srcA = Amean + (size_t)(mTile + r) * C; kcol = kb; }
        else        { srcA = Aself + (size_t)(mTile + r) * C; kcol = kb - C; }
        v16bf a;
#pragma unroll
        for (int j = 0; j < 8; ++j) a[j] = (__bf16)srcA[kcol + j];
#pragma unroll
        for (int j = 0; j < 8; ++j) a[8 + j] = (__bf16)srcA[kcol + 16 + j];
        *(v16bf*)(sA + (size_t)p * 16) = a;
    }
    __syncthreads();

    // ---- phase 2: per-wave WMMA over K ----
    const int ntIdx = blockIdx.y * 8 + wave;          // global column-tile index
    v8f acc = {};
#pragma unroll
    for (int kt = 0; kt < KT; ++kt) {
        v16bf a = *(const v16bf*)(sA + ((size_t)(kt * 32 + lane)) * 16);
        const __bf16* bp = Wp + (((size_t)kt * NT + ntIdx) * 32 + lane) * 16;
        v16bf b = *(const v16bf*)bp;
        acc = __builtin_amdgcn_wmma_f32_16x16x32_bf16(
            /*neg_a=*/false, a, /*neg_b=*/false, b,
            /*c_mod=*/(short)0, acc, /*reuse_a=*/false, /*reuse_b=*/false);
    }

    // ---- epilogue: bias + ReLU, C/D layout (lane<16: M=v; lane>=16: M=v+8) -
    int col = ntIdx * 16 + (lane & 15);
    float bb = bias[col];
#pragma unroll
    for (int v = 0; v < 8; ++v) {
        int m = v + ((lane >> 4) ? 8 : 0);
        float val = acc[v] + bb;
        out[(size_t)(mTile + m) * COUT + col] = val > 0.0f ? val : 0.0f;
    }
}

// ---------------- global mean pool: pooled[batch[i],:] += h3[i,:] ----------
__global__ void pool_kernel(const int* __restrict__ batch, const float* __restrict__ h3,
                            float* __restrict__ pooled, float* __restrict__ gcnt) {
    unsigned t = blockIdx.x * blockDim.x + threadIdx.x;   // N*128 threads
    unsigned i = t >> 7;
    unsigned c = t & 127u;
    if (i >= NNODES) return;
    unsigned g = (unsigned)batch[i];
    atomicAdd(&pooled[(size_t)g * HDIM + c], h3[t]);
    if (c == 0) atomicAdd(&gcnt[g], 1.0f);
}

// ---------------- final FC: out[g,cls] -------------------------------------
__global__ void fc_kernel(const float* __restrict__ pooled, const float* __restrict__ gcnt,
                          const float* __restrict__ fcw, const float* __restrict__ fcb,
                          float* __restrict__ out) {
    int t = blockIdx.x * blockDim.x + threadIdx.x;        // G*CLS threads
    if (t >= GRAPHS * NCLS) return;
    int g  = t / NCLS;
    int cl = t % NCLS;
    float inv = 1.0f / fmaxf(gcnt[g], 1.0f);
    float s = fcb[cl];
    for (int c = 0; c < HDIM; ++c)
        s += pooled[(size_t)g * HDIM + c] * inv * fcw[c * NCLS + cl];
    out[t] = s;
}

// ---------------------------------------------------------------------------
extern "C" void kernel_launch(void* const* d_in, const int* in_sizes, int n_in,
                              void* d_out, int out_size, void* d_ws, size_t ws_size,
                              hipStream_t stream) {
    const float* x    = (const float*)d_in[0];
    const int*   eidx = (const int*)d_in[1];
    const int*   src  = eidx;
    const int*   dst  = eidx + NEDGES;
    const int*   batch = (const int*)d_in[2];
    const float* W1l = (const float*)d_in[3];
    const float* b1  = (const float*)d_in[4];
    const float* W1r = (const float*)d_in[5];
    const float* W2l = (const float*)d_in[6];
    const float* b2  = (const float*)d_in[7];
    const float* W2r = (const float*)d_in[8];
    const float* W3l = (const float*)d_in[9];
    const float* b3  = (const float*)d_in[10];
    const float* W3r = (const float*)d_in[11];
    const float* fcw = (const float*)d_in[12];
    const float* fcb = (const float*)d_in[13];
    float* out = (float*)d_out;

    // ---- workspace carve-up (all offsets 512B aligned) ----
    char* ws = (char*)d_ws;
    size_t off = 0;
    auto carve = [&](size_t bytes) {
        char* p = ws + off;
        off += (bytes + 511) & ~(size_t)511;
        return p;
    };
    float*  cnt    = (float*)carve((size_t)NNODES * 4);            // in-degree
    float*  agg1   = (float*)carve((size_t)NNODES * 4);            // scalar agg
    float*  agg    = (float*)carve((size_t)NNODES * 256 * 4);      // mean buffer (reused)
    float*  h1     = (float*)carve((size_t)NNODES * 128 * 4);      // layer1 out / layer3 out
    float*  h2     = (float*)carve((size_t)NNODES * 256 * 4);      // layer2 out
    __bf16* Wp2    = (__bf16*)carve((size_t)256 * 256 * 2);        // packed [W2l;W2r]
    __bf16* Wp3    = (__bf16*)carve((size_t)512 * 128 * 2);        // packed [W3l;W3r]
    float*  pooled = (float*)carve((size_t)GRAPHS * HDIM * 4);
    float*  gcnt   = (float*)carve((size_t)GRAPHS * 4);
    float*  h3     = h1;                                           // alias (h1 dead by then)
    (void)ws_size; (void)in_sizes; (void)n_in; (void)out_size;

    const int TB = 256;

    // ---- degree + layer-1 scalar aggregation ----
    hipMemsetAsync(cnt,  0, (size_t)NNODES * 4, stream);
    hipMemsetAsync(agg1, 0, (size_t)NNODES * 4, stream);
    deg_agg1_kernel<<<(NEDGES + TB - 1) / TB, TB, 0, stream>>>(src, dst, x, cnt, agg1);

    // ---- layer 1 (rank-1) ----
    layer1_kernel<<<(NNODES * 128 + TB - 1) / TB, TB, 0, stream>>>(x, agg1, cnt, W1l, b1, W1r, h1);

    // ---- layer 2: aggregate h1, mean, pack weights, WMMA GEMM -> h2 ----
    hipMemsetAsync(agg, 0, (size_t)NNODES * 128 * 4, stream);
    edge_agg_kernel<7><<<((size_t)NEDGES * 32 + TB - 1) / TB, TB, 0, stream>>>(src, dst, h1, agg);
    mean_scale_kernel<7><<<(NNODES * 128 + TB - 1) / TB, TB, 0, stream>>>(agg, cnt);
    pack_w_kernel<128, 256><<<(2 * 128 * 256 + TB - 1) / TB, TB, 0, stream>>>(W2l, W2r, Wp2);
    {
        dim3 grid(NNODES / 16, 256 / 128);   // 6250 x 2, 8 waves/block = 8 col tiles
        sage_gemm_kernel<128, 256><<<grid, TB, 0, stream>>>(agg, h1, Wp2, b2, h2);
    }

    // ---- layer 3: aggregate h2, mean, pack weights, WMMA GEMM -> h3 ----
    hipMemsetAsync(agg, 0, (size_t)NNODES * 256 * 4, stream);
    edge_agg_kernel<8><<<((size_t)NEDGES * 64 + TB - 1) / TB, TB, 0, stream>>>(src, dst, h2, agg);
    mean_scale_kernel<8><<<(NNODES * 256 + TB - 1) / TB, TB, 0, stream>>>(agg, cnt);
    pack_w_kernel<256, 128><<<(2 * 256 * 128 + TB - 1) / TB, TB, 0, stream>>>(W3l, W3r, Wp3);
    {
        dim3 grid(NNODES / 16, 128 / 128);   // 6250 x 1
        sage_gemm_kernel<256, 128><<<grid, TB, 0, stream>>>(agg, h2, Wp3, b3, h3);
    }

    // ---- global mean pool + FC ----
    hipMemsetAsync(pooled, 0, (size_t)GRAPHS * HDIM * 4, stream);
    hipMemsetAsync(gcnt,   0, (size_t)GRAPHS * 4, stream);
    pool_kernel<<<(NNODES * 128 + TB - 1) / TB, TB, 0, stream>>>(batch, h3, pooled, gcnt);
    fc_kernel<<<(GRAPHS * NCLS + TB - 1) / TB, TB, 0, stream>>>(pooled, gcnt, fcw, fcb, out);
}
